// SeqParallelAttn_25409026523795
// MI455X (gfx1250) — compile-verified, matrix operand order checked by reading, not run on hardware
//
#include <hip/hip_runtime.h>
#include <hip/hip_bf16.h>

// ---------------------------------------------------------------------------
// Types for CDNA5 WMMA (wave32): v_wmma_f32_16x16x32_bf16
// ---------------------------------------------------------------------------
typedef __attribute__((ext_vector_type(16))) __bf16 v16bf;
typedef __attribute__((ext_vector_type(8)))  float  v8f;
typedef __attribute__((ext_vector_type(4)))  __bf16 v4bf;
typedef __attribute__((ext_vector_type(4)))  int    v4i;
typedef __attribute__((ext_vector_type(4)))  unsigned int u32x4;
typedef __attribute__((ext_vector_type(8)))  int    i32x8;
typedef __attribute__((ext_vector_type(4)))  int    i32x4;

// ---------------------------------------------------------------------------
// Async global->LDS staging (GLOBAL_LOAD_ASYNC_TO_LDS_B128, ASYNCcnt).
// Probe (round 2): builtins exist; params are v4i addrspace(1)*/(3)*.
// ---------------------------------------------------------------------------
#if defined(__has_builtin)
#if __has_builtin(__builtin_amdgcn_global_load_async_to_lds_b128) && \
    __has_builtin(__builtin_amdgcn_s_wait_asynccnt)
#define USE_ASYNC 1
#endif
#if __has_builtin(__builtin_amdgcn_tensor_load_to_lds) && \
    __has_builtin(__builtin_amdgcn_s_wait_tensorcnt)
#define USE_TDM 1
#endif
#endif
#ifndef USE_ASYNC
#define USE_ASYNC 0
#endif
#ifndef USE_TDM
#define USE_TDM 0
#endif

#if USE_ASYNC
#define WAIT_ASYNC(n) __builtin_amdgcn_s_wait_asynccnt(n)
#else
#define WAIT_ASYNC(n) do {} while (0)
#endif
#if USE_TDM
#define WAIT_TENSOR(n) __builtin_amdgcn_s_wait_tensorcnt(n)
#else
#define WAIT_TENSOR(n) do {} while (0)
#endif

// 16-byte global->LDS copy. Flat LDS addresses carry the LDS byte offset in
// the low 32 bits (aperture in the high bits), so truncation yields a valid
// addrspace(3) pointer.
__device__ __forceinline__ void copy16_g2l(const __bf16* src, __bf16* dst) {
#if USE_ASYNC
  typedef __attribute__((address_space(1))) v4i gv4i_t;
  typedef __attribute__((address_space(3))) v4i lv4i_t;
  __builtin_amdgcn_global_load_async_to_lds_b128(
      (gv4i_t*)(uintptr_t)src,
      (lv4i_t*)(unsigned)(uintptr_t)dst,
      0, 0);
#else
  *(uint4*)dst = *(const uint4*)src;
#endif
}

#if USE_TDM
// TDM 2D tile load: rows x cols bf16, global row stride in elements,
// packed contiguously into LDS (cols elements per row).  Descriptor per
// CDNA5 ISA ch.8: group0 {count=1, lds_addr, global_addr, type=2},
// group1 {data_size=1 (2B), tensor dims (large: no OOB), tile dims, stride}.
__device__ __forceinline__ void tdm_load_2d(const __bf16* gsrc, __bf16* ldst,
                                            int rows, int cols,
                                            int rowstride) {
  const unsigned lds_off = (unsigned)(uintptr_t)ldst;  // low 32b = LDS offset
  const unsigned long long ga = (unsigned long long)(uintptr_t)gsrc;
  u32x4 g0;
  g0[0] = 1u;                                    // count=1, user descriptor
  g0[1] = lds_off;                               // lds_addr
  g0[2] = (unsigned)ga;                          // global_addr[31:0]
  g0[3] = (unsigned)(ga >> 32) | 0x80000000u;    // addr[56:32] | type=2
  i32x8 g1;
  const unsigned td0 = 1u << 20, td1 = 1u << 20; // tensor dims: no OOB
  g1[0] = 0x00010000;                            // data_size=1 (2 bytes)
  g1[1] = (int)((td0 & 0xFFFFu) << 16);          // tensor_dim0[15:0]
  g1[2] = (int)(((td0 >> 16) & 0xFFFFu) | ((td1 & 0xFFFFu) << 16));
  g1[3] = (int)(((td1 >> 16) & 0xFFFFu) | ((unsigned)cols << 16)); // tile_dim0
  g1[4] = rows;                                  // tile_dim1; tile_dim2=0
  g1[5] = rowstride;                             // tensor_dim0_stride[31:0]
  g1[6] = 0;                                     // stride hi / dim1_stride
  g1[7] = 0;
  const i32x4 z = {0, 0, 0, 0};
#if __clang_major__ >= 23
  const i32x8 z8 = {0, 0, 0, 0, 0, 0, 0, 0};
  __builtin_amdgcn_tensor_load_to_lds(g0, g1, z, z, z8, 0);
#else
  __builtin_amdgcn_tensor_load_to_lds(g0, g1, z, z, 0);
#endif
}
#endif

union FragU { uint4 u[2]; v16bf v; };

// Load a 16x32 bf16 WMMA operand fragment from a K-major matrix.
// ISA 7.12.2 layout: lane L holds row (L&15); halves 0..7 = K[8*(L>>4) ..],
// halves 8..15 = K[16 + 8*(L>>4) ..].  Two 16B loads per lane.
__device__ __forceinline__ v16bf frag_kmajor(const __bf16* base, int ld) {
  const int lane = threadIdx.x & 31;
  const __bf16* p = base + (lane & 15) * ld + ((lane >> 4) << 3);
  FragU f;
  f.u[0] = *(const uint4*)(p);
  f.u[1] = *(const uint4*)(p + 16);
  return f.v;
}

__device__ __forceinline__ v8f wmma_bf16(v16bf a, v16bf b, v8f c) {
  return __builtin_amdgcn_wmma_f32_16x16x32_bf16(false, a, false, b,
                                                 (short)0, c, false, false);
}

// ---------------------------------------------------------------------------
// fp32 -> bf16 cast (vectorized: float4 in, 4x bf16 out)
// ---------------------------------------------------------------------------
__global__ void cast_f32_bf16(const float4* __restrict__ in,
                              v4bf* __restrict__ out, int n4) {
  for (int i = blockIdx.x * blockDim.x + threadIdx.x; i < n4;
       i += gridDim.x * blockDim.x) {
    const float4 v = in[i];
    v4bf o;
    o.x = (__bf16)v.x; o.y = (__bf16)v.y;
    o.z = (__bf16)v.z; o.w = (__bf16)v.w;
    out[i] = o;
  }
}

// ---------------------------------------------------------------------------
// Generic 2048x2048x2048 bf16 GEMM:  out = A(2048xK) * Bw(2048xK)^T
// Block 128x128, 256 threads = 8 waves, wave tile 32x64 (8 C tiles).
// Double-buffered 128x64 LDS tiles staged by the Tensor Data Mover
// (one descriptor per tile, issued by wave 0, TENSORcnt-tracked), with
// async-to-LDS / sync fallbacks.
// MODE 0: bf16 row-major | MODE 1: bf16 (B,H,S,hd) | MODE 2: bf16 Vt hd-major
// MODE 3: fp32 row-major
// ---------------------------------------------------------------------------
template <int MODE>
__global__ __launch_bounds__(256) void gemm2048(const __bf16* __restrict__ A,
                                                const __bf16* __restrict__ Bw,
                                                void* __restrict__ outp) {
  constexpr int KD = 2048;
  __shared__ __bf16 As[2][128 * 64];
  __shared__ __bf16 Bs[2][128 * 64];
  const int tid  = threadIdx.x;
  const int lane = tid & 31;
  const int wave = tid >> 5;
  const int wr   = wave & 3;   // rows wr*32
  const int wc   = wave >> 2;  // cols wc*64
  const int R0   = (blockIdx.x >> 4) * 128;
  const int C0   = (blockIdx.x & 15) * 128;

  const v8f zf = {0.f, 0.f, 0.f, 0.f, 0.f, 0.f, 0.f, 0.f};
  v8f acc[8];
#pragma unroll
  for (int t = 0; t < 8; ++t) acc[t] = zf;

  // stage one 128x64 tile pair
  auto stage = [&](int buf, int k0) {
#if USE_TDM
    if (wave == 0) {
      tdm_load_2d(&A[(size_t)R0 * KD + k0], &As[buf][0], 128, 64, KD);
      tdm_load_2d(&Bw[(size_t)C0 * KD + k0], &Bs[buf][0], 128, 64, KD);
    }
#else
    for (int c = tid; c < 1024; c += 256) {
      const int row = c >> 3, off = (c & 7) << 3;
      copy16_g2l(&A[(size_t)(R0 + row) * KD + k0 + off],
                 &As[buf][row * 64 + off]);
      copy16_g2l(&Bw[(size_t)(C0 + row) * KD + k0 + off],
                 &Bs[buf][row * 64 + off]);
    }
#endif
  };

  stage(0, 0);
  for (int t = 0; t < 32; ++t) {
    const int cur = t & 1;
    if (t + 1 < 32) {
      stage(cur ^ 1, (t + 1) * 64);
#if USE_TDM
      WAIT_TENSOR(2);  // previous stage's 2 descriptors retired (in order)
#else
      WAIT_ASYNC(8);   // previous stage's 8 per-wave async ops are done
#endif
    } else {
#if USE_TDM
      WAIT_TENSOR(0);
#else
      WAIT_ASYNC(0);
#endif
    }
    __syncthreads();

#pragma unroll
    for (int kk = 0; kk < 64; kk += 32) {
      v16bf af[2], bfr[4];
#pragma unroll
      for (int i = 0; i < 2; ++i)
        af[i] = frag_kmajor(&As[cur][(wr * 32 + i * 16) * 64 + kk], 64);
#pragma unroll
      for (int j = 0; j < 4; ++j)
        bfr[j] = frag_kmajor(&Bs[cur][(wc * 64 + j * 16) * 64 + kk], 64);
#pragma unroll
      for (int i = 0; i < 2; ++i)
#pragma unroll
        for (int j = 0; j < 4; ++j)
          acc[i * 4 + j] = wmma_bf16(af[i], bfr[j], acc[i * 4 + j]);
    }
    __syncthreads();  // buffer reuse fence before next prefetch
  }

  // epilogue (C layout: VGPR e -> row e + 8*(lane>>4), col lane&15)
#pragma unroll
  for (int i = 0; i < 2; ++i)
#pragma unroll
    for (int j = 0; j < 4; ++j) {
      if (MODE == 2) {
        // e-run is contiguous in s: pack 8 bf16 into one b128 store
        union { uint4 u; __bf16 h[8]; } pk;
#pragma unroll
        for (int e = 0; e < 8; ++e) pk.h[e] = (__bf16)acc[i * 4 + j][e];
        const int grow0 = R0 + wr * 32 + i * 16 + ((lane >> 4) << 3);
        const int gcol  = C0 + wc * 64 + j * 16 + (lane & 15);
        const int b = grow0 >> 10, s = grow0 & 1023;
        const int h = gcol >> 7, d = gcol & 127;
        *(uint4*)&((__bf16*)outp)[(size_t)d * 32768 + (b * 16 + h) * 1024 +
                                  s] = pk.u;
      } else {
#pragma unroll
        for (int e = 0; e < 8; ++e) {
          const int grow = R0 + wr * 32 + i * 16 + e + ((lane >> 4) << 3);
          const int gcol = C0 + wc * 64 + j * 16 + (lane & 15);
          const float v = acc[i * 4 + j][e];
          if (MODE == 0) {
            ((__bf16*)outp)[(size_t)grow * 2048 + gcol] = (__bf16)v;
          } else if (MODE == 1) {
            const int b = grow >> 10, s = grow & 1023;
            const int h = gcol >> 7, d = gcol & 127;
            ((__bf16*)outp)[(size_t)((b * 16 + h) * 1024 + s) * 128 + d] =
                (__bf16)v;
          } else {
            ((float*)outp)[(size_t)grow * 2048 + gcol] = v;
          }
        }
      }
    }
}

// ---------------------------------------------------------------------------
// Flash attention over the simulated all-gather.
// k_global(b,h,kg,d) = Kflat[(g*4096*128 + kg*128 + d) mod 2^22], g = b*16+h.
// Grid: (B*H)*(S/64) = 512 blocks, 128 threads = 4 waves; wave owns 16 query
// rows; 64-key tiles double-buffered via async global->LDS copies (ASYNCcnt).
// ---------------------------------------------------------------------------
__global__ __launch_bounds__(128) void attn_kernel(
    const __bf16* __restrict__ Q,   // (B*S, 2048) row-major
    const __bf16* __restrict__ Kl,  // (B,H,S,hd) flat, 2^22 elements
    const __bf16* __restrict__ Vt,  // (128, 32768): Vt[d][(b*16+h)*1024+s]
    __bf16* __restrict__ Out) {     // (B*S, 2048) row-major
  __shared__ __bf16 Kt[2][64 * 128];    // key tile, key-major (ld=128)
  __shared__ __bf16 Vts[2][128 * 64];   // value tile, d-major (ld=64)
  __shared__ __bf16 Pb[4][16 * 64];     // per-wave P transpose scratch

  const int tid  = threadIdx.x;
  const int lane = tid & 31;
  const int wave = tid >> 5;
  const int g  = blockIdx.x >> 4;   // b*16 + h
  const int qb = blockIdx.x & 15;
  const int b = g >> 4, h = g & 15;
  const int qrow0 = qb * 64 + wave * 16;
  const float scale = 0.08838834764831845f;  // 1/sqrt(128)

  // preload Q fragments (16 rows x 128 K)
  v16bf qf[4];
#pragma unroll
  for (int kk = 0; kk < 4; ++kk)
    qf[kk] = frag_kmajor(
        Q + (size_t)(b * 1024 + qrow0) * 2048 + h * 128 + kk * 32, 2048);

  const v8f zf = {0.f, 0.f, 0.f, 0.f, 0.f, 0.f, 0.f, 0.f};
  v8f acc[8];
#pragma unroll
  for (int c = 0; c < 8; ++c) acc[c] = zf;
  float m_i[8], l_i[8];
#pragma unroll
  for (int e = 0; e < 8; ++e) { m_i[e] = -3.4e38f; l_i[e] = 0.0f; }

  // stage one K/V tile pair (16 async b128 instructions per wave)
  auto stage = [&](int buf, int j) {
    const size_t kbase = ((size_t)g * 524288u + (size_t)j * 8192u) & 4194303u;
    for (int c = tid; c < 1024; c += 128)
      copy16_g2l(&Kl[kbase + (size_t)c * 8], &Kt[buf][c * 8]);
    const int R0 = (g * 4096 + j * 64) & 32767;
    for (int c = tid; c < 1024; c += 128) {
      const int d = c >> 3, off = (c & 7) << 3;
      copy16_g2l(&Vt[(size_t)d * 32768 + R0 + off], &Vts[buf][d * 64 + off]);
    }
  };

  stage(0, 0);
  for (int j = 0; j < 64; ++j) {
    const int cur = j & 1;
    if (j + 1 < 64) {
      stage(cur ^ 1, j + 1);
      WAIT_ASYNC(16);
    } else {
      WAIT_ASYNC(0);
    }
    __syncthreads();

    // S = Q K^T : 4 score tiles, contraction 128 = 4 WMMA steps
    v8f sc[4];
#pragma unroll
    for (int t = 0; t < 4; ++t) {
      sc[t] = zf;
#pragma unroll
      for (int kk = 0; kk < 4; ++kk)
        sc[t] = wmma_bf16(
            qf[kk], frag_kmajor(&Kt[cur][(t * 16) * 128 + kk * 32], 128),
            sc[t]);
    }

    // online softmax: row max / exp / row sum (16-lane xor reductions)
    float alpha[8];
#pragma unroll
    for (int e = 0; e < 8; ++e) {
      float v0 = sc[0][e] * scale, v1 = sc[1][e] * scale;
      float v2 = sc[2][e] * scale, v3 = sc[3][e] * scale;
      sc[0][e] = v0; sc[1][e] = v1; sc[2][e] = v2; sc[3][e] = v3;
      float m = fmaxf(fmaxf(v0, v1), fmaxf(v2, v3));
#pragma unroll
      for (int msk = 1; msk < 16; msk <<= 1)
        m = fmaxf(m, __shfl_xor(m, msk, 32));
      const float mn = fmaxf(m_i[e], m);
      alpha[e] = __expf(m_i[e] - mn);
      m_i[e] = mn;
      float s0 = 0.f;
#pragma unroll
      for (int t = 0; t < 4; ++t) {
        const float p = __expf(sc[t][e] - mn);
        sc[t][e] = p;
        s0 += p;
      }
#pragma unroll
      for (int msk = 1; msk < 16; msk <<= 1)
        s0 += __shfl_xor(s0, msk, 32);
      l_i[e] = l_i[e] * alpha[e] + s0;
    }

    // P: C-layout -> A-layout through per-wave LDS (intra-wave only)
#pragma unroll
    for (int t = 0; t < 4; ++t)
#pragma unroll
      for (int e = 0; e < 8; ++e) {
        const int r = e + ((lane >> 4) << 3);
        Pb[wave][r * 64 + t * 16 + (lane & 15)] = (__bf16)sc[t][e];
      }

    // rescale accumulator
#pragma unroll
    for (int c = 0; c < 8; ++c)
#pragma unroll
      for (int e = 0; e < 8; ++e) acc[c][e] *= alpha[e];

    // O += P @ V : contraction 64 keys = 2 WMMA steps per 16-col tile
    v16bf pf[2];
#pragma unroll
    for (int kk = 0; kk < 2; ++kk)
      pf[kk] = frag_kmajor(&Pb[wave][kk * 32], 64);
#pragma unroll
    for (int c = 0; c < 8; ++c)
#pragma unroll
      for (int kk = 0; kk < 2; ++kk)
        acc[c] = wmma_bf16(
            pf[kk], frag_kmajor(&Vts[cur][(c * 16) * 64 + kk * 32], 64),
            acc[c]);

    __syncthreads();  // tile buffer reuse fence before next prefetch
  }

  // normalize and store to (B,S,D) row-major for the Wo GEMM
#pragma unroll
  for (int e = 0; e < 8; ++e) {
    const float rl = 1.0f / l_i[e];
    const int r = qrow0 + e + ((lane >> 4) << 3);
#pragma unroll
    for (int c = 0; c < 8; ++c) {
      Out[(size_t)(b * 1024 + r) * 2048 + h * 128 + c * 16 + (lane & 15)] =
          (__bf16)(acc[c][e] * rl);
    }
  }
}

// ---------------------------------------------------------------------------
// Host launcher
// ---------------------------------------------------------------------------
extern "C" void kernel_launch(void* const* d_in, const int* in_sizes, int n_in,
                              void* d_out, int out_size, void* d_ws,
                              size_t ws_size, hipStream_t stream) {
  (void)in_sizes; (void)n_in; (void)out_size; (void)ws_size;
  const float* x  = (const float*)d_in[0];
  const float* Wq = (const float*)d_in[1];
  const float* Wk = (const float*)d_in[2];
  const float* Wv = (const float*)d_in[3];
  const float* Wo = (const float*)d_in[4];

  char* ws = (char*)d_ws;
  const size_t MB = 1ull << 20;
  __bf16* xb  = (__bf16*)(ws + 0 * MB);
  __bf16* wqb = (__bf16*)(ws + 8 * MB);
  __bf16* wkb = (__bf16*)(ws + 16 * MB);
  __bf16* wvb = (__bf16*)(ws + 24 * MB);
  __bf16* wob = (__bf16*)(ws + 32 * MB);
  __bf16* Qb  = (__bf16*)(ws + 40 * MB);
  __bf16* Kb  = (__bf16*)(ws + 48 * MB);
  __bf16* Vtb = (__bf16*)(ws + 56 * MB);
  __bf16* Ab  = (__bf16*)(ws + 64 * MB);

  const int N4 = (2048 * 2048) / 4;
  cast_f32_bf16<<<512, 256, 0, stream>>>((const float4*)x,  (v4bf*)xb,  N4);
  cast_f32_bf16<<<512, 256, 0, stream>>>((const float4*)Wq, (v4bf*)wqb, N4);
  cast_f32_bf16<<<512, 256, 0, stream>>>((const float4*)Wk, (v4bf*)wkb, N4);
  cast_f32_bf16<<<512, 256, 0, stream>>>((const float4*)Wv, (v4bf*)wvb, N4);
  cast_f32_bf16<<<512, 256, 0, stream>>>((const float4*)Wo, (v4bf*)wob, N4);

  gemm2048<0><<<256, 256, 0, stream>>>(xb, wqb, Qb);   // Q row-major
  gemm2048<1><<<256, 256, 0, stream>>>(xb, wkb, Kb);   // K (B,H,S,hd)
  gemm2048<2><<<256, 256, 0, stream>>>(xb, wvb, Vtb);  // V transposed

  attn_kernel<<<512, 128, 0, stream>>>(Qb, Kb, Vtb, Ab);

  gemm2048<3><<<256, 256, 0, stream>>>(Ab, wob, d_out);  // fp32 out
}
